// BaseMPNN_61486751809987
// MI455X (gfx1250) — matrix operand order repaired and checked by reading, not compiled, not run on hardware
//
#include <hip/hip_runtime.h>
#include <stdint.h>

#define NNODES 100000
#define NEDGES 1600000
#define HID    64
#define NITERS 3

typedef __attribute__((ext_vector_type(16))) __bf16 v16bf;
typedef __attribute__((ext_vector_type(8)))  float  v8f;

union Frag8 {
  uint32_t u[8];
  v16bf    bf;
};

// ---- fp32 -> bf16 helpers (round-to-nearest-even); only used in cold kernels ----
__device__ __forceinline__ uint32_t pack_bf16(float lo, float hi) {
  uint32_t a = __float_as_uint(lo);
  uint32_t b = __float_as_uint(hi);
  a += 0x7FFFu + ((a >> 16) & 1u);
  b += 0x7FFFu + ((b >> 16) & 1u);
  return (a >> 16) | (b & 0xFFFF0000u);
}
__device__ __forceinline__ uint16_t f32_to_bf16_rne(float f) {
  uint32_t u = __float_as_uint(f);
  u += 0x7FFFu + ((u >> 16) & 1u);
  return (uint16_t)(u >> 16);
}
__device__ __forceinline__ float bf16_to_f32(uint16_t v) {
  return __uint_as_float(((uint32_t)v) << 16);
}

__device__ __forceinline__ v8f wmma_bf16(const Frag8& A, const Frag8& B, v8f C) {
  return __builtin_amdgcn_wmma_f32_16x16x32_bf16(
      /*neg_a=*/false, A.bf, /*neg_b=*/false, B.bf,
      /*c_mod=*/(short)0, C, /*reuse_a=*/false, /*reuse_b=*/false);
}

// 16x32 bf16 A-fragment from a bf16 row of length 64: pure b128 loads, no VALU.
// Lane<16 holds K = kc*32 + {0..7, 16..23}; lane>=16 holds K = kc*32 + {8..15, 24..31}.
__device__ __forceinline__ void load_afrag_bf16(const uint16_t* __restrict__ row,
                                                int kc, int lane, Frag8& A) {
  int base = kc * 32 + ((lane & 16) ? 8 : 0);
  uint4 lo = *(const uint4*)(row + base);        // K = base .. base+7
  uint4 hi = *(const uint4*)(row + base + 16);   // K = base+16 .. base+23
  A.u[0] = lo.x; A.u[1] = lo.y; A.u[2] = lo.z; A.u[3] = lo.w;
  A.u[4] = hi.x; A.u[5] = hi.y; A.u[6] = hi.z; A.u[7] = hi.w;
}

// 16x32 bf16 A-fragment built from an fp32 row (agg); RNE pack (GRU only).
__device__ __forceinline__ void build_afrag_f32(const float* __restrict__ row,
                                                int kc, int lane, Frag8& A) {
  int base = kc * 32 + ((lane & 16) ? 8 : 0);
  float4 a0 = *(const float4*)(row + base);
  float4 a1 = *(const float4*)(row + base + 4);
  float4 b0 = *(const float4*)(row + base + 16);
  float4 b1 = *(const float4*)(row + base + 20);
  A.u[0] = pack_bf16(a0.x, a0.y);  A.u[1] = pack_bf16(a0.z, a0.w);
  A.u[2] = pack_bf16(a1.x, a1.y);  A.u[3] = pack_bf16(a1.z, a1.w);
  A.u[4] = pack_bf16(b0.x, b0.y);  A.u[5] = pack_bf16(b0.z, b0.w);
  A.u[6] = pack_bf16(b1.x, b1.y);  A.u[7] = pack_bf16(b1.z, b1.w);
}

// Load pre-swizzled 32x16 bf16 B-fragment: 8 dwords per lane, contiguous (2x b128).
__device__ __forceinline__ void load_bfrag(const uint32_t* __restrict__ frag,
                                           int kc, int nc, int ntiles_n, int lane, Frag8& B) {
  const uint32_t* p = frag + ((size_t)((kc * ntiles_n + nc) * 32 + lane) << 3);
  uint4 lo = *(const uint4*)p;
  uint4 hi = *(const uint4*)(p + 4);
  B.u[0] = lo.x; B.u[1] = lo.y; B.u[2] = lo.z; B.u[3] = lo.w;
  B.u[4] = hi.x; B.u[5] = hi.y; B.u[6] = hi.z; B.u[7] = hi.w;
}

// ---- Weight pre-swizzle: row-major fp32 [K][N] -> bf16 B-fragment layout ----
__global__ void prep_wfrag_kernel(const float* __restrict__ W, uint32_t* __restrict__ frag,
                                  int K, int N) {
  int idx = blockIdx.x * blockDim.x + threadIdx.x;
  int total = (K / 32) * (N / 16) * 32 * 8;
  if (idx >= total) return;
  int v    = idx & 7;
  int lane = (idx >> 3) & 31;
  int tile = idx >> 8;
  int NT = N / 16;
  int kc = tile / NT, nc = tile % NT;
  int n = nc * 16 + (lane & 15);
  int k = kc * 32 + ((lane & 16) ? 8 : 0) + 2 * v + ((v & 4) ? 8 : 0);
  frag[idx] = pack_bf16(W[(size_t)k * N + n], W[(size_t)(k + 1) * N + n]);
}

__global__ void zero_kernel(float* __restrict__ p, int n) {
  int i = blockIdx.x * blockDim.x + threadIdx.x;
  int stride = gridDim.x * blockDim.x;
  for (; i < n; i += stride) p[i] = 0.0f;
}

// ---- Edge message + scatter-add: one wave per 16-edge tile ----
// msg = relu(h[senders] @ W_msg + b_msg); agg[receivers] += msg (f32 atomics)
__global__ void __launch_bounds__(128, 1)
edge_msg_kernel(const uint16_t* __restrict__ hbf,
                const int* __restrict__ senders,
                const int* __restrict__ receivers,
                const uint32_t* __restrict__ wmsg_frag,
                const float* __restrict__ b_msg,
                float* __restrict__ agg,
                int n_tiles) {
  int lane   = threadIdx.x & 31;
  int wave   = (blockIdx.x * blockDim.x + threadIdx.x) >> 5;
  int nwaves = (gridDim.x * blockDim.x) >> 5;

  // W_msg fragments: 2 K-chunks x 4 N-tiles, held in registers for all tiles.
  Frag8 B[2][4];
#pragma unroll
  for (int kc = 0; kc < 2; ++kc)
#pragma unroll
    for (int nc = 0; nc < 4; ++nc)
      load_bfrag(wmsg_frag, kc, nc, 4, lane, B[kc][nc]);

  int col = lane & 15;
  float bias[4];
#pragma unroll
  for (int nc = 0; nc < 4; ++nc) bias[nc] = b_msg[nc * 16 + col];

  const v8f vzero = {0.f, 0.f, 0.f, 0.f, 0.f, 0.f, 0.f, 0.f};
  uint32_t mbase = (lane & 16) ? 8u : 0u;

  for (int t = wave; t < n_tiles; t += nwaves) {
    uint32_t ebase = (uint32_t)t * 16u;
    int tn = t + nwaves;
    if (tn < n_tiles) {                       // hide next tile's index latency
      __builtin_prefetch(senders + (uint32_t)tn * 16u, 0, 0);
      __builtin_prefetch(receivers + (uint32_t)tn * 16u, 0, 0);
    }
    uint32_t srow = (uint32_t)senders[ebase + col];   // row of A this lane feeds
    const uint16_t* row = hbf + (size_t)srow * HID;

    v8f acc[4] = {vzero, vzero, vzero, vzero};
#pragma unroll
    for (int kc = 0; kc < 2; ++kc) {
      Frag8 A;
      load_afrag_bf16(row, kc, lane, A);      // 2x b128, zero VALU
#pragma unroll
      for (int nc = 0; nc < 4; ++nc) acc[nc] = wmma_bf16(A, B[kc][nc], acc[nc]);
    }

    // receivers for the 8 rows this lane holds in D (32-bit element offsets)
    uint32_t roff[8];
    int4 r0 = *(const int4*)(receivers + ebase + mbase);
    int4 r1 = *(const int4*)(receivers + ebase + mbase + 4);
    roff[0] = (uint32_t)r0.x << 6; roff[1] = (uint32_t)r0.y << 6;
    roff[2] = (uint32_t)r0.z << 6; roff[3] = (uint32_t)r0.w << 6;
    roff[4] = (uint32_t)r1.x << 6; roff[5] = (uint32_t)r1.y << 6;
    roff[6] = (uint32_t)r1.z << 6; roff[7] = (uint32_t)r1.w << 6;

#pragma unroll
    for (int nc = 0; nc < 4; ++nc) {
      uint32_t n = (uint32_t)(nc * 16) + (uint32_t)col;
#pragma unroll
      for (int v = 0; v < 8; ++v) {
        float val = fmaxf(acc[nc][v] + bias[nc], 0.0f);
        unsafeAtomicAdd(agg + (roff[v] + n), val);   // saddr + 32-bit voffset
      }
    }
  }
}

// ---- GRU update: one wave per 16-node tile; bf16 state in, bf16 state out ----
__global__ void __launch_bounds__(128, 1)
gru_kernel(const uint16_t* __restrict__ hin_bf,
           const float* __restrict__ agg,
           const uint32_t* __restrict__ wi_frag,
           const uint32_t* __restrict__ wh_frag,
           const float* __restrict__ b_i,
           const float* __restrict__ b_h,
           uint16_t* __restrict__ hout_bf,
           int n_tiles) {
  int lane = threadIdx.x & 31;
  int wave = (blockIdx.x * blockDim.x + threadIdx.x) >> 5;
  if (wave >= n_tiles) return;                // wave-uniform exit, EXEC stays all-1

  int nbase = wave * 16;
  int col   = lane & 15;
  int mbase = (lane & 16) ? 8 : 0;

  const uint16_t* hrow = hin_bf + (size_t)(nbase + col) * HID;
  const float*    arow = agg    + (size_t)(nbase + col) * HID;

  Frag8 Ah[2], Aa[2];
#pragma unroll
  for (int kc = 0; kc < 2; ++kc) {
    load_afrag_bf16(hrow, kc, lane, Ah[kc]);
    build_afrag_f32(arow, kc, lane, Aa[kc]);
  }

  const v8f vzero = {0.f, 0.f, 0.f, 0.f, 0.f, 0.f, 0.f, 0.f};

#pragma unroll 1
  for (int dc = 0; dc < 4; ++dc) {            // 16-column output chunk (rolled)
    v8f ir = vzero, iz = vzero, in_ = vzero;
    v8f hr = vzero, hz = vzero, hn = vzero;
#pragma unroll
    for (int kc = 0; kc < 2; ++kc) {
      Frag8 Bir, Biz, Bin, Bhr, Bhz, Bhn;
      load_bfrag(wi_frag, kc, dc,     12, lane, Bir);
      load_bfrag(wi_frag, kc, dc + 4, 12, lane, Biz);
      load_bfrag(wi_frag, kc, dc + 8, 12, lane, Bin);
      load_bfrag(wh_frag, kc, dc,     12, lane, Bhr);
      load_bfrag(wh_frag, kc, dc + 4, 12, lane, Bhz);
      load_bfrag(wh_frag, kc, dc + 8, 12, lane, Bhn);
      ir  = wmma_bf16(Aa[kc], Bir, ir);
      iz  = wmma_bf16(Aa[kc], Biz, iz);
      in_ = wmma_bf16(Aa[kc], Bin, in_);
      hr  = wmma_bf16(Ah[kc], Bhr, hr);
      hz  = wmma_bf16(Ah[kc], Bhz, hz);
      hn  = wmma_bf16(Ah[kc], Bhn, hn);
    }
    int d = dc * 16 + col;
    float bir = b_i[d], biz = b_i[HID + d], bin = b_i[2 * HID + d];
    float bhr = b_h[d], bhz = b_h[HID + d], bhn = b_h[2 * HID + d];
#pragma unroll
    for (int v = 0; v < 8; ++v) {
      int node = nbase + mbase + v;           // D row -> node
      float hold = bf16_to_f32(hin_bf[(size_t)node * HID + d]);
      float r = 1.0f / (1.0f + __expf(-((ir[v] + bir) + (hr[v] + bhr))));
      float z = 1.0f / (1.0f + __expf(-((iz[v] + biz) + (hz[v] + bhz))));
      float nv = tanhf((in_[v] + bin) + r * (hn[v] + bhn));
      hout_bf[(size_t)node * HID + d] = f32_to_bf16_rne((1.0f - z) * nv + z * hold);
    }
  }
}

// ---- Readout ----
__global__ void node_sum_kernel(const uint16_t* __restrict__ hbf, float* __restrict__ g,
                                int n_nodes) {
  int tid = blockIdx.x * blockDim.x + threadIdx.x;
  int d = tid & (HID - 1);
  int node = tid >> 6;
  int stride = (gridDim.x * blockDim.x) >> 6;
  float s = 0.0f;
  for (; node < n_nodes; node += stride) s += bf16_to_f32(hbf[(size_t)node * HID + d]);
  unsafeAtomicAdd(&g[d], s);
}

__global__ void readout_kernel(const float* __restrict__ g,
                               const float* __restrict__ W_ro,
                               const float* __restrict__ b_ro,
                               float* __restrict__ out) {
  int j = threadIdx.x;  // 64 threads
  float s = b_ro[j];
#pragma unroll 8
  for (int k = 0; k < HID; ++k) s += g[k] * W_ro[k * HID + j];
  out[j] = s;
}

extern "C" void kernel_launch(void* const* d_in, const int* in_sizes, int n_in,
                              void* d_out, int out_size, void* d_ws, size_t ws_size,
                              hipStream_t stream) {
  (void)in_sizes; (void)n_in; (void)out_size; (void)ws_size;
  const int*   senders   = (const int*)d_in[0];
  const int*   receivers = (const int*)d_in[1];
  const float* W_msg     = (const float*)d_in[2];
  const float* b_msg     = (const float*)d_in[3];
  const float* W_i       = (const float*)d_in[4];
  const float* W_h       = (const float*)d_in[5];
  const float* b_i       = (const float*)d_in[6];
  const float* b_h       = (const float*)d_in[7];
  const float* W_ro      = (const float*)d_in[8];
  const float* b_ro      = (const float*)d_in[9];
  float* out = (float*)d_out;

  // Workspace carve-out (256B aligned): bf16 h ping-pong, f32 agg, swizzled weights, g.
  size_t off = 0;
  char* base = (char*)d_ws;
  auto carve = [&](size_t bytes) -> void* {
    void* p = base + off;
    off += (bytes + 255) & ~(size_t)255;
    return p;
  };
  uint16_t* hbf0      = (uint16_t*)carve((size_t)NNODES * HID * 2);
  uint16_t* hbf1      = (uint16_t*)carve((size_t)NNODES * HID * 2);
  float*    agg       = (float*)carve((size_t)NNODES * HID * 4);
  uint32_t* wmsg_frag = (uint32_t*)carve(2 * 4 * 32 * 8 * 4);    // 64x64
  uint32_t* wi_frag   = (uint32_t*)carve(2 * 12 * 32 * 8 * 4);   // 64x192
  uint32_t* wh_frag   = (uint32_t*)carve(2 * 12 * 32 * 8 * 4);   // 64x192
  float*    g         = (float*)carve(HID * 4);

  // Pre-swizzle weights into WMMA B-fragment layout (bf16 packed).
  prep_wfrag_kernel<<<(2048 + 255) / 256, 256, 0, stream>>>(W_msg, wmsg_frag, 64, 64);
  prep_wfrag_kernel<<<(6144 + 255) / 256, 256, 0, stream>>>(W_i, wi_frag, 64, 192);
  prep_wfrag_kernel<<<(6144 + 255) / 256, 256, 0, stream>>>(W_h, wh_frag, 64, 192);

  const int hElems = NNODES * HID;
  zero_kernel<<<2048, 256, 0, stream>>>((float*)hbf0, hElems / 2);  // bf16 zeros

  uint16_t* hbuf[2] = {hbf0, hbf1};
  const int edge_tiles = NEDGES / 16;   // 100000
  const int node_tiles = NNODES / 16;   // 6250
  const int gru_blocks = (node_tiles + 3) / 4;  // 4 waves per 128-thread block

  for (int it = 0; it < NITERS; ++it) {
    uint16_t* hcur = hbuf[it & 1];
    uint16_t* hnxt = hbuf[(it + 1) & 1];
    zero_kernel<<<4096, 256, 0, stream>>>(agg, hElems);
    edge_msg_kernel<<<8192, 128, 0, stream>>>(hcur, senders, receivers,
                                              wmsg_frag, b_msg, agg, edge_tiles);
    gru_kernel<<<gru_blocks, 128, 0, stream>>>(hcur, agg, wi_frag, wh_frag,
                                               b_i, b_h, hnxt, node_tiles);
  }

  zero_kernel<<<1, 64, 0, stream>>>(g, HID);
  node_sum_kernel<<<1024, 256, 0, stream>>>(hbuf[NITERS & 1], g, NNODES);
  readout_kernel<<<1, 64, 0, stream>>>(g, W_ro, b_ro, out);
}
